// InductionHeadAttention_1778116461293
// MI455X (gfx1250) — matrix operand order repaired and checked by reading, not compiled, not run on hardware
//
#include <hip/hip_runtime.h>
#include <hip/hip_bf16.h>
#include <stdint.h>

typedef unsigned short u16;
typedef __bf16 bf16;
typedef bf16         v16bf __attribute__((ext_vector_type(16)));
typedef float        v8f   __attribute__((ext_vector_type(8)));
typedef unsigned int v8u   __attribute__((ext_vector_type(8)));

__device__ __forceinline__ u16 f2bf(float f) {
  unsigned u = __float_as_uint(f);
  unsigned r = u + 0x7fffu + ((u >> 16) & 1u);   // round-to-nearest-even
  return (u16)(r >> 16);
}
__device__ __forceinline__ float bf2f(u16 h) {
  return __uint_as_float(((unsigned)h) << 16);
}

// ---------------------------------------------------------------------------
// CDNA5 async global->LDS copy (ASYNCcnt-tracked), 16 bytes per lane.
// dsaddr = LDS byte offset (low 32 bits of generic LDS pointer).
// ---------------------------------------------------------------------------
__device__ __forceinline__ void async_b128(unsigned ldsaddr, const u16* g) {
  asm volatile("global_load_async_to_lds_b128 %0, %1, off"
               :: "v"(ldsaddr), "v"(g) : "memory");
}
__device__ __forceinline__ void wait_async_le4() {
  asm volatile("s_wait_asynccnt 0x4" ::: "memory");
}
__device__ __forceinline__ void wait_async_le0() {
  asm volatile("s_wait_asynccnt 0x0" ::: "memory");
}

// ---------------------------------------------------------------------------
// fp32 -> bf16 flat conversion (emb)
// ---------------------------------------------------------------------------
__global__ void k_f32_to_bf16(const float* __restrict__ in, u16* __restrict__ out, int n) {
  int i = blockIdx.x * blockDim.x + threadIdx.x;
  if (i < n) out[i] = f2bf(in[i]);
}

// ---------------------------------------------------------------------------
// Fused transpose + convert: out[n][k] = bf16(in[k][n]); K,N multiples of 32.
// ---------------------------------------------------------------------------
__global__ __launch_bounds__(256) void k_transpose_cvt(const float* __restrict__ in,
                                                       u16* __restrict__ out, int K, int N) {
  __shared__ float tile[32][33];
  const int kb = blockIdx.y * 32, nb = blockIdx.x * 32;
  const int tx = threadIdx.x & 31, ty = threadIdx.x >> 5;   // 32 x 8
  #pragma unroll
  for (int i = ty; i < 32; i += 8)
    tile[i][tx] = in[(size_t)(kb + i) * N + nb + tx];
  __syncthreads();
  #pragma unroll
  for (int i = ty; i < 32; i += 8)
    out[(size_t)(nb + i) * K + kb + tx] = f2bf(tile[tx][i]);
}

// ---------------------------------------------------------------------------
// Batched bf16 transpose: out[b][d][s] = in[b*S+s][d]; S,D multiples of 32.
// ---------------------------------------------------------------------------
__global__ __launch_bounds__(256) void k_transpose_bf16(const u16* __restrict__ in,
                                                        u16* __restrict__ out, int S, int D) {
  __shared__ u16 tile[32][34];
  const int b = blockIdx.z;
  const int sb = blockIdx.x * 32, db = blockIdx.y * 32;
  const int tx = threadIdx.x & 31, ty = threadIdx.x >> 5;
  #pragma unroll
  for (int i = ty; i < 32; i += 8)
    tile[i][tx] = in[((size_t)b * S + sb + i) * D + db + tx];
  __syncthreads();
  #pragma unroll
  for (int i = ty; i < 32; i += 8)
    out[((size_t)b * D + db + i) * S + sb + tx] = tile[tx][i];
}

// ---------------------------------------------------------------------------
// Build X[b,s,:] = concat(emb[tok[b,s]], pos[s,:])  (bf16)
// ---------------------------------------------------------------------------
__global__ void k_build_x(const int* __restrict__ tok, const float* __restrict__ emb,
                          const float* __restrict__ pos, u16* __restrict__ X,
                          int Sl, int Vn, int Dm) {
  const int row = blockIdx.x;          // b*S + s
  const int s = row % Sl;
  const int t = tok[row];
  u16* xr = X + (size_t)row * Dm;
  for (int i = threadIdx.x; i < Vn; i += blockDim.x) xr[i] = f2bf(emb[t * Vn + i]);
  const float* pr = pos + (size_t)s * Sl;
  for (int j = threadIdx.x; j < Sl; j += blockDim.x) xr[Vn + j] = f2bf(pr[j]);
}

// ---------------------------------------------------------------------------
// bf16 WMMA GEMM, B stored [N,K] (C = A * B^T), async double-buffered LDS.
// MODE 0: C bf16 = acc
// MODE 1: C f32  = acc * alpha * (*alphaPtr), causal mask (col > row -> -1e30)
// MODE 2: C bf16 = acc + R (bf16 residual)
// MODE 3: C f32  = acc * alpha * (*alphaPtr)
// Block: 256 threads (8 waves), 128x128 tile, K-step 32.
// Out-of-range tile rows are source-clamped: garbage only reaches C rows/cols
// that the epilogue never stores.
// ---------------------------------------------------------------------------
template<int MODE>
__global__ __launch_bounds__(256) void k_gemm(
    const u16* __restrict__ A, size_t sAb, int lda,
    const u16* __restrict__ B, size_t sBb, int ldb,
    void* __restrict__ C, size_t sCb, int ldc,
    const u16* __restrict__ R, size_t sRb, int ldr,
    int M, int N, int K, float alpha, const float* __restrict__ alphaPtr)
{
  __shared__ u16 As[2][128][40];   // row stride 80B (multiple of 16B)
  __shared__ u16 Bs[2][128][40];   // Bs[buf][n][k]

  const int t    = threadIdx.x;
  const int lane = t & 31, lo = lane & 15, hi = lane >> 4;
  const int w    = t >> 5;
  const int wm   = w & 3;       // 32-row group
  const int wn   = w >> 2;      // 64-col group
  const int rowBase = blockIdx.y * 128;
  const int colBase = blockIdx.x * 128;
  const size_t offA = (size_t)blockIdx.z * sAb;
  const size_t offB = (size_t)blockIdx.z * sBb;
  const size_t offC = (size_t)blockIdx.z * sCb;
  const size_t offR = (size_t)blockIdx.z * sRb;

  v8f acc[2][4] = {};

  const int ar = t >> 1;            // 0..127
  const int ac = (t & 1) << 4;      // 0 or 16
  const int arow = min(rowBase + ar, M - 1);   // clamp OOB staging rows
  const int brow = min(colBase + ar, N - 1);
  const u16* aptr = A + offA + (size_t)arow * lda + ac;
  const u16* bptr = B + offB + (size_t)brow * ldb + ac;

  const int nk = K >> 5;

  auto stage = [&](int kb, int buf) {
    const unsigned la = (unsigned)(uintptr_t)&As[buf][ar][ac];
    const unsigned lb = (unsigned)(uintptr_t)&Bs[buf][ar][ac];
    const u16* ga = aptr + kb;
    const u16* gb = bptr + kb;
    async_b128(la,       ga);
    async_b128(la + 16u, ga + 8);
    async_b128(lb,       gb);
    async_b128(lb + 16u, gb + 8);
  };

  stage(0, 0);
  if (nk > 1) stage(32, 1);

  for (int k = 0; k < nk; ++k) {
    if (k + 1 < nk) wait_async_le4(); else wait_async_le0();
    __syncthreads();                 // stage-k data visible to all waves
    const int buf = k & 1;

    // fragments per ISA 16-bit A (16x32) / B (32x16) lane layouts
    v16bf afrag[2];
    #pragma unroll
    for (int i = 0; i < 2; ++i) {
      const int r = wm * 32 + i * 16 + lo;
      v8u av;
      #pragma unroll
      for (int v = 0; v < 8; ++v) {
        const int kk = ((v < 4) ? (v << 1) : (16 + ((v - 4) << 1))) + (hi ? 8 : 0);
        av[v] = *(const unsigned*)&As[buf][r][kk];
      }
      afrag[i] = __builtin_bit_cast(v16bf, av);
    }
    v16bf bfrag[4];
    #pragma unroll
    for (int j = 0; j < 4; ++j) {
      const int c = wn * 64 + j * 16 + lo;
      v8u bv;
      #pragma unroll
      for (int v = 0; v < 8; ++v) {
        const int kk = ((v < 4) ? (v << 1) : (16 + ((v - 4) << 1))) + (hi ? 8 : 0);
        bv[v] = *(const unsigned*)&Bs[buf][c][kk];
      }
      bfrag[j] = __builtin_bit_cast(v16bf, bv);
    }

    #pragma unroll
    for (int i = 0; i < 2; ++i)
      #pragma unroll
      for (int j = 0; j < 4; ++j)
        acc[i][j] = __builtin_amdgcn_wmma_f32_16x16x32_bf16(
            false, afrag[i], false, bfrag[j], (short)0, acc[i][j], false, false);

    __syncthreads();                 // all waves done reading buf
    if (k + 2 < nk) stage((k + 2) << 5, buf);
  }

  float aEff = alpha;
  if (alphaPtr) aEff *= *alphaPtr;

  // C/D layout: VGPR r -> M = r + hi*8, N = lo (per 16x16 tile)
  #pragma unroll
  for (int i = 0; i < 2; ++i) {
    #pragma unroll
    for (int j = 0; j < 4; ++j) {
      #pragma unroll
      for (int r = 0; r < 8; ++r) {
        const int row = rowBase + wm * 32 + i * 16 + r + (hi ? 8 : 0);
        const int col = colBase + wn * 64 + j * 16 + lo;
        if (row >= M) continue;
        float val = acc[i][j][r];
        if (MODE == 0) {
          ((u16*)C)[offC + (size_t)row * ldc + col] = f2bf(val);
        } else if (MODE == 1) {
          val *= aEff;
          if (col > row) val = -1.0e30f;           // causal mask
          ((float*)C)[offC + (size_t)row * ldc + col] = val;
        } else if (MODE == 2) {
          val += bf2f(R[offR + (size_t)row * ldr + col]);
          ((u16*)C)[offC + (size_t)row * ldc + col] = f2bf(val);
        } else {
          ((float*)C)[offC + (size_t)row * ldc + col] = val * aEff;
        }
      }
    }
  }
}

// ---------------------------------------------------------------------------
// Row softmax: P[row,:] = softmax(S[row,:]), n <= 2048, 256 threads
// ---------------------------------------------------------------------------
__global__ __launch_bounds__(256) void k_softmax_rows(const float* __restrict__ S,
                                                      u16* __restrict__ P, int n) {
  __shared__ float red[256];
  const size_t row = blockIdx.x;
  const float* sp = S + row * (size_t)n;
  const int t = threadIdx.x;
  float ev[8];
  float mx = -3.0e38f;
  for (int c = t, i = 0; c < n; c += 256, ++i) { float v = sp[c]; ev[i] = v; mx = fmaxf(mx, v); }
  red[t] = mx; __syncthreads();
  for (int s = 128; s > 0; s >>= 1) { if (t < s) red[t] = fmaxf(red[t], red[t + s]); __syncthreads(); }
  mx = red[0]; __syncthreads();
  float sum = 0.f;
  for (int c = t, i = 0; c < n; c += 256, ++i) { float e = __expf(ev[i] - mx); ev[i] = e; sum += e; }
  red[t] = sum; __syncthreads();
  for (int s = 128; s > 0; s >>= 1) { if (t < s) red[t] += red[t + s]; __syncthreads(); }
  const float inv = 1.0f / red[0];
  u16* pp = P + row * (size_t)n;
  for (int c = t, i = 0; c < n; c += 256, ++i) pp[c] = f2bf(ev[i] * inv);
}

// ---------------------------------------------------------------------------
// q2[b,:] = Z1[b, S-1, :] @ WQ2  (WQ2 passed pre-transposed [V, D])
// ---------------------------------------------------------------------------
__global__ __launch_bounds__(128) void k_q2(const u16* __restrict__ Z1,
                                            const u16* __restrict__ WQ2t,
                                            float* __restrict__ q2, int Sl, int Dm, int Vn) {
  const int b = blockIdx.x, v = threadIdx.x;
  const u16* zr = Z1 + ((size_t)b * Sl + (Sl - 1)) * Dm;
  const u16* wr = WQ2t + (size_t)v * Dm;
  float acc = 0.f;
  for (int d = 0; d < Dm; ++d) acc += bf2f(zr[d]) * bf2f(wr[d]);
  q2[b * Vn + v] = acc;
}

// ---------------------------------------------------------------------------
// Stage-2 single-query attention per batch: Y2 = softmax(beta2*scale2*q2.K2^T).V2
// ---------------------------------------------------------------------------
__global__ __launch_bounds__(128) void k_stage2(const float* __restrict__ q2,
    const u16* __restrict__ K2, const u16* __restrict__ V2,
    const float* __restrict__ beta2p, float scale2,
    float* __restrict__ Y2f, u16* __restrict__ Y2b, int Sl, int Vn) {
  __shared__ float s2[2048];
  __shared__ float red[128];
  __shared__ float qs[128];
  const int b = blockIdx.x, t = threadIdx.x;
  qs[t] = q2[b * Vn + t];
  __syncthreads();
  const float beta = beta2p[0] * scale2;
  float mx = -3.0e38f;
  for (int s = t; s < Sl; s += 128) {
    const u16* kr = K2 + ((size_t)b * Sl + s) * Vn;
    float d = 0.f;
    for (int v = 0; v < 128; ++v) d += qs[v] * bf2f(kr[v]);
    d *= beta;
    s2[s] = d;
    mx = fmaxf(mx, d);
  }
  red[t] = mx; __syncthreads();
  for (int s = 64; s > 0; s >>= 1) { if (t < s) red[t] = fmaxf(red[t], red[t + s]); __syncthreads(); }
  mx = red[0]; __syncthreads();
  float sum = 0.f;
  for (int s = t; s < Sl; s += 128) { float e = __expf(s2[s] - mx); s2[s] = e; sum += e; }
  red[t] = sum; __syncthreads();
  for (int s = 64; s > 0; s >>= 1) { if (t < s) red[t] += red[t + s]; __syncthreads(); }
  const float inv = 1.0f / red[0];
  __syncthreads();
  float acc = 0.f;
  for (int s = 0; s < Sl; ++s) acc += s2[s] * bf2f(V2[((size_t)b * Sl + s) * Vn + t]);
  acc *= inv;
  Y2f[b * Vn + t] = acc;
  Y2b[b * Vn + t] = f2bf(acc);
}

// ---------------------------------------------------------------------------
extern "C" void kernel_launch(void* const* d_in, const int* in_sizes, int n_in,
                              void* d_out, int out_size, void* d_ws, size_t ws_size,
                              hipStream_t stream) {
  (void)in_sizes; (void)n_in; (void)out_size; (void)ws_size;
  const int Bn = 64, Sl = 2048, Vn = 128, Dm = Vn + Sl;   // 2176
  const int BS = Bn * Sl;                                  // 131072

  const int*   tok   = (const int*)  d_in[0];
  const float* emb   = (const float*)d_in[1];
  const float* pos   = (const float*)d_in[2];
  const float* WQ1   = (const float*)d_in[3];
  const float* WK1   = (const float*)d_in[4];
  const float* WV1   = (const float*)d_in[5];
  const float* WQ2   = (const float*)d_in[6];
  const float* WK2   = (const float*)d_in[7];
  const float* WV2   = (const float*)d_in[8];
  const float* beta1 = (const float*)d_in[9];
  const float* beta2 = (const float*)d_in[10];
  const float* betaO = (const float*)d_in[11];

  char* base = (char*)d_ws;
  size_t off = 0;
  auto alloc = [&](size_t bytes) -> char* {
    char* p = base + off;
    off = (off + bytes + 255) & ~(size_t)255;
    return p;
  };
  u16*   embB  = (u16*)  alloc((size_t)Vn * Vn * 2);
  u16*   WQ1t  = (u16*)  alloc((size_t)Dm * Sl * 2);   // [Sl, Dm]
  u16*   WK1t  = (u16*)  alloc((size_t)Dm * Sl * 2);   // [Sl, Dm]
  u16*   WV1t  = (u16*)  alloc((size_t)Dm * Dm * 2);   // [Dm, Dm]
  u16*   WQ2t  = (u16*)  alloc((size_t)Dm * Vn * 2);   // [Vn, Dm]
  u16*   WK2t  = (u16*)  alloc((size_t)Dm * Vn * 2);   // [Vn, Dm]
  u16*   WV2t  = (u16*)  alloc((size_t)Dm * Vn * 2);   // [Vn, Dm]
  u16*   Xb    = (u16*)  alloc((size_t)BS * Dm * 2);
  u16*   Q1    = (u16*)  alloc((size_t)BS * Sl * 2);
  u16*   K1    = (u16*)  alloc((size_t)BS * Sl * 2);
  u16*   V1    = (u16*)  alloc((size_t)BS * Dm * 2);   // [b*S+s][d]
  u16*   V1t   = (u16*)  alloc((size_t)BS * Dm * 2);   // [b][d][s]
  float* S1    = (float*)alloc((size_t)Bn * Sl * Sl * 4);
  u16*   P1    = (u16*)  alloc((size_t)Bn * Sl * Sl * 2);
  u16*   Z1    = (u16*)  alloc((size_t)BS * Dm * 2);
  float* q2f   = (float*)alloc((size_t)Bn * Vn * 4);
  u16*   K2    = (u16*)  alloc((size_t)BS * Vn * 2);
  u16*   V2    = (u16*)  alloc((size_t)BS * Vn * 2);
  float* Y2f   = (float*)alloc((size_t)Bn * Vn * 4);
  u16*   Y2b   = (u16*)  alloc((size_t)Bn * Vn * 2);

  // weights: fused transpose + bf16 convert  (out[n][k] = in[k][n])
  k_f32_to_bf16<<<dim3((Vn * Vn + 255) / 256), 256, 0, stream>>>(emb, embB, Vn * Vn);
  k_transpose_cvt<<<dim3(Sl / 32, Dm / 32), 256, 0, stream>>>(WQ1, WQ1t, Dm, Sl);
  k_transpose_cvt<<<dim3(Sl / 32, Dm / 32), 256, 0, stream>>>(WK1, WK1t, Dm, Sl);
  k_transpose_cvt<<<dim3(Dm / 32, Dm / 32), 256, 0, stream>>>(WV1, WV1t, Dm, Dm);
  k_transpose_cvt<<<dim3(Vn / 32, Dm / 32), 256, 0, stream>>>(WQ2, WQ2t, Dm, Vn);
  k_transpose_cvt<<<dim3(Vn / 32, Dm / 32), 256, 0, stream>>>(WK2, WK2t, Dm, Vn);
  k_transpose_cvt<<<dim3(Vn / 32, Dm / 32), 256, 0, stream>>>(WV2, WV2t, Dm, Vn);

  k_build_x<<<BS, 256, 0, stream>>>(tok, emb, pos, Xb, Sl, Vn, Dm);

  // stage-1 projections (B = W^T stored [N,K])
  k_gemm<0><<<dim3(Sl / 128, BS / 128, 1), 256, 0, stream>>>(
      Xb, 0, Dm, WQ1t, 0, Dm, Q1, 0, Sl, nullptr, 0, 0, BS, Sl, Dm, 1.f, nullptr);
  k_gemm<0><<<dim3(Sl / 128, BS / 128, 1), 256, 0, stream>>>(
      Xb, 0, Dm, WK1t, 0, Dm, K1, 0, Sl, nullptr, 0, 0, BS, Sl, Dm, 1.f, nullptr);
  k_gemm<0><<<dim3(Dm / 128, BS / 128, 1), 256, 0, stream>>>(
      Xb, 0, Dm, WV1t, 0, Dm, V1, 0, Dm, nullptr, 0, 0, BS, Dm, Dm, 1.f, nullptr);

  // V1^T per batch for the P @ V1 GEMM
  k_transpose_bf16<<<dim3(Sl / 32, Dm / 32, Bn), 256, 0, stream>>>(V1, V1t, Sl, Dm);

  // S1 = beta1 * scale1 * Q1 @ K1^T (causal), batched over 64
  const float scale1 = (float)(((double)Dm / Sl) * ((double)Dm / Sl));
  k_gemm<1><<<dim3(Sl / 128, Sl / 128, Bn), 256, 0, stream>>>(
      Q1, (size_t)Sl * Sl, Sl, K1, (size_t)Sl * Sl, Sl, S1, (size_t)Sl * Sl, Sl,
      nullptr, 0, 0, Sl, Sl, Sl, scale1, beta1);

  k_softmax_rows<<<BS, 256, 0, stream>>>(S1, P1, Sl);

  // Z1 = X + P1 @ V1  (B = V1^T [Dm, Sl] per batch)
  k_gemm<2><<<dim3(Dm / 128, Sl / 128, Bn), 256, 0, stream>>>(
      P1, (size_t)Sl * Sl, Sl, V1t, (size_t)Sl * Dm, Sl, Z1, (size_t)Sl * Dm, Dm,
      Xb, (size_t)Sl * Dm, Dm, Sl, Dm, Sl, 1.f, nullptr);

  // stage 2
  k_q2<<<Bn, 128, 0, stream>>>(Z1, WQ2t, q2f, Sl, Dm, Vn);
  k_gemm<0><<<dim3(Vn / 128, BS / 128, 1), 256, 0, stream>>>(
      Z1, 0, Dm, WK2t, 0, Dm, K2, 0, Vn, nullptr, 0, 0, BS, Vn, Dm, 1.f, nullptr);
  k_gemm<0><<<dim3(Vn / 128, BS / 128, 1), 256, 0, stream>>>(
      Z1, 0, Dm, WV2t, 0, Dm, V2, 0, Vn, nullptr, 0, 0, BS, Vn, Dm, 1.f, nullptr);

  const float scale2 = (float)(((double)Dm / Vn) * ((double)Dm / Vn));  // 289
  k_stage2<<<Bn, 128, 0, stream>>>(q2f, K2, V2, beta2, scale2, Y2f, Y2b, Sl, Vn);

  // logits = beta_out * (D/V)^1.5 * Y2 @ emb^T  (emb row-major is already [N,K])
  const float oscale = 70.0927947f;  // 17^1.5
  k_gemm<3><<<dim3(Vn / 128, 1, 1), 256, 0, stream>>>(
      Y2b, 0, Vn, embB, 0, Vn, (float*)d_out, 0, Vn,
      nullptr, 0, 0, Bn, Vn, Vn, oscale, betaO);
}
